// MambaTracker_31903017075093
// MI455X (gfx1250) — matrix operand order repaired and checked by reading, not compiled, not run on hardware
//
#include <hip/hip_runtime.h>
#include <hip/hip_bf16.h>
#include <stdint.h>

// ---------------------------------------------------------------------------
// MambaTracker forward for gfx1250 (MI455X).
// All dense matmuls use v_wmma_f32_16x16x32_f16 (wave32, 16x16 tile / wave).
// Conv stem is algebraically fused with the 14x14 mean pool (9 shifted block
// sums), so the 822MB conv activation is never materialized.
// ---------------------------------------------------------------------------

typedef __attribute__((ext_vector_type(16))) _Float16 half16;
typedef __attribute__((ext_vector_type(8)))  _Float16 half8;
typedef __attribute__((ext_vector_type(8)))  float    float8;

#define NT   256              // tokens (16x16 pooled grid)
#define NB   32               // image batch == Mamba scan length L
#define ROWS (NT * NB)        // 8192 GEMM rows, row = t*32 + l
#define DM   128
#define DSTATE 8

// ---------------- generic 16x16-tile WMMA accumulator ----------------------
// C_tile = A[m0:m0+16, :K] * W[n0:n0+16, :K]^T ; A,W row-major f16.
__device__ __forceinline__ float8 wmma_tile_f16(
    const _Float16* __restrict__ A, const _Float16* __restrict__ W,
    int m0, int n0, int K, int lda, int ldb)
{
  float8 acc;
#pragma unroll
  for (int i = 0; i < 8; ++i) acc[i] = 0.0f;
  const int lane = threadIdx.x & 31;
  const int hs   = lane >> 4;                 // half-wave select
  const _Float16* arow = A + (size_t)(m0 + (lane & 15)) * lda;
  const _Float16* brow = W + (size_t)(n0 + (lane & 15)) * ldb;
  for (int k0 = 0; k0 < K; k0 += 32) {
    // A 16x32 fragment: lane half 0 -> K 0..7 & 16..23 ; half 1 -> 8..15 & 24..31
    half8 a0 = *(const half8*)(arow + k0 + hs * 8);
    half8 a1 = *(const half8*)(arow + k0 + 16 + hs * 8);
    // B 32x16 fragment: lane half 0 -> K 0..7 & 16..23 of col n ; half 1 -> 8..15 & 24..31
    half8 b0 = *(const half8*)(brow + k0 + hs * 8);
    half8 b1 = *(const half8*)(brow + k0 + 16 + hs * 8);
    half16 av = __builtin_shufflevector(a0, a1, 0,1,2,3,4,5,6,7,8,9,10,11,12,13,14,15);
    half16 bv = __builtin_shufflevector(b0, b1, 0,1,2,3,4,5,6,7,8,9,10,11,12,13,14,15);
    acc = __builtin_amdgcn_wmma_f32_16x16x32_f16(false, av, false, bv,
                                                 (short)0, acc, false, false);
  }
  return acc;
}

// ---------------- K0: pack / pad all weights to f16 ------------------------
// segments: in_proj 256x128 | x_proj pad 32x128 | dt_proj pad 128x32 |
//           out_proj 128x128 | heads pad 80x128 (det 0..4, id 5..68, zero 69..79)
__global__ void k_pack_weights(const float* __restrict__ in_proj_w,
                               const float* __restrict__ x_proj_w,
                               const float* __restrict__ dt_proj_w,
                               const float* __restrict__ out_proj_w,
                               const float* __restrict__ det_w,
                               const float* __restrict__ id_w,
                               _Float16* w_in, _Float16* w_xp, _Float16* w_dt,
                               _Float16* w_out, _Float16* w_hd)
{
  int i = blockIdx.x * 256 + threadIdx.x;
  if (i < 32768) { w_in[i] = (_Float16)in_proj_w[i]; return; }
  i -= 32768;
  if (i < 4096) { int n = i >> 7, k = i & 127;
    w_xp[i] = (_Float16)((n < 24) ? x_proj_w[n * 128 + k] : 0.0f); return; }
  i -= 4096;
  if (i < 4096) { int n = i >> 5, k = i & 31;
    w_dt[i] = (_Float16)((k < 8) ? dt_proj_w[n * 8 + k] : 0.0f); return; }
  i -= 4096;
  if (i < 16384) { w_out[i] = (_Float16)out_proj_w[i]; return; }
  i -= 16384;
  if (i < 10240) { int n = i >> 7, k = i & 127;
    float v = (n < 5) ? det_w[n * 128 + k] : ((n < 69) ? id_w[(n - 5) * 128 + k] : 0.0f);
    w_hd[i] = (_Float16)v; }
}

// ---------------- K1: fused conv stem (3x3 pad1) + 14x14 mean pool ---------
// block = (t, l): t = ph*16+pw token, l = image index. 256 threads.
__global__ void k_stem_pool(const float* __restrict__ x,
                            const float* __restrict__ w1,
                            const float* __restrict__ b1,
                            _Float16* __restrict__ u16)
{
  const int t = blockIdx.x >> 5;       // token 0..255
  const int bimg = blockIdx.x & 31;    // image 0..31
  const int ph = t >> 4, pw = t & 15;
  __shared__ float Spart[27][8];
  __shared__ float S[27];
  const int tid = threadIdx.x;
  if (tid < 216) {
    const int s = tid >> 3, j = tid & 7;          // s = ci*9 + kh*3 + kw
    const int ci = s / 9, kh = (s / 3) % 3, kw = s % 3;
    const float* xp = x + ((size_t)bimg * 3 + ci) * 224 * 224;
    float acc = 0.0f;
    for (int e = j; e < 196; e += 8) {
      int hh = ph * 14 + (e / 14) + kh - 1;
      int ww = pw * 14 + (e % 14) + kw - 1;
      if (hh >= 0 && hh < 224 && ww >= 0 && ww < 224) acc += xp[hh * 224 + ww];
    }
    Spart[s][j] = acc;
  }
  __syncthreads();
  if (tid < 27) {
    float a = 0.0f;
#pragma unroll
    for (int j = 0; j < 8; ++j) a += Spart[tid][j];
    S[tid] = a;
  }
  __syncthreads();
  if (tid < 128) {
    const int c = tid;
    const float* wc = w1 + c * 27;
    float s = 0.0f;
#pragma unroll
    for (int k = 0; k < 27; ++k) s += wc[k] * S[k];
    float v = b1[c] + s * (1.0f / 196.0f);
    u16[((size_t)t * 32 + bimg) * 128 + c] = (_Float16)v;
  }
}

// ---------------- K2: xz = u @ in_proj^T  (8192x256, K=128) ----------------
__global__ void k_gemm_inproj(const _Float16* __restrict__ A,
                              const _Float16* __restrict__ W,
                              float* __restrict__ xz)
{
  const int tile = blockIdx.x * 8 + (threadIdx.x >> 5);
  const int m0 = (tile >> 4) << 4, n0 = (tile & 15) << 4;
  float8 acc = wmma_tile_f16(A, W, m0, n0, 128, 128, 128);
  const int lane = threadIdx.x & 31;
  const int col = n0 + (lane & 15), rbase = m0 + (lane >> 4) * 8;
#pragma unroll
  for (int r = 0; r < 8; ++r) xz[(size_t)(rbase + r) * 256 + col] = acc[r];
}

// ---------------- K3: causal depthwise conv (len-2 along l) + SiLU ---------
__global__ void k_dwconv_silu(const float* __restrict__ xz,
                              const float* __restrict__ cw,
                              const float* __restrict__ cb,
                              float* __restrict__ xs, _Float16* __restrict__ xs16)
{
  const int g = blockIdx.x * 256 + threadIdx.x;   // < 8192*128
  const int d = g & 127, row = g >> 7, l = row & 31;
  float xc = xz[(size_t)row * 256 + d];
  float xp = (l > 0) ? xz[(size_t)(row - 1) * 256 + d] : 0.0f;
  float v = cw[d * 2] * xp + cw[d * 2 + 1] * xc + cb[d];
  v = v / (1.0f + __expf(-v));
  xs[(size_t)row * 128 + d] = v;
  xs16[(size_t)row * 128 + d] = (_Float16)v;
}

// ---------------- K4: x_dbl = xs @ x_proj^T (8192x32pad, K=128) ------------
__global__ void k_gemm_xproj(const _Float16* __restrict__ A,
                             const _Float16* __restrict__ W,
                             float* __restrict__ xdbl, _Float16* __restrict__ xdbl16)
{
  const int tile = blockIdx.x * 8 + (threadIdx.x >> 5);
  const int m0 = (tile >> 1) << 4, n0 = (tile & 1) << 4;
  float8 acc = wmma_tile_f16(A, W, m0, n0, 128, 128, 128);
  const int lane = threadIdx.x & 31;
  const int col = n0 + (lane & 15), rbase = m0 + (lane >> 4) * 8;
#pragma unroll
  for (int r = 0; r < 8; ++r) {
    size_t idx = (size_t)(rbase + r) * 32 + col;
    xdbl[idx] = acc[r];
    xdbl16[idx] = (_Float16)acc[r];
  }
}

// ---------------- K5: dt = softplus(dt_raw @ dt_proj^T + b) (K=32 padded) --
__global__ void k_gemm_dtproj(const _Float16* __restrict__ A,
                              const _Float16* __restrict__ W,
                              const float* __restrict__ dtb,
                              float* __restrict__ dt)
{
  const int tile = blockIdx.x * 8 + (threadIdx.x >> 5);
  const int m0 = (tile >> 3) << 4, n0 = (tile & 7) << 4;
  float8 acc = wmma_tile_f16(A, W, m0, n0, 32, 32, 32);
  const int lane = threadIdx.x & 31;
  const int col = n0 + (lane & 15), rbase = m0 + (lane >> 4) * 8;
#pragma unroll
  for (int r = 0; r < 8; ++r) {
    float v = acc[r] + dtb[col];
    v = (v > 20.0f) ? v : __logf(1.0f + __expf(v));
    dt[(size_t)(rbase + r) * 128 + col] = v;
  }
}

// ---------------- K6: selective scan (L=32, 8 states in regs) + gate -------
__global__ void k_scan(const float* __restrict__ dt, const float* __restrict__ xs,
                       const float* __restrict__ xdbl, const float* __restrict__ xz,
                       const float* __restrict__ A_log, const float* __restrict__ Dp,
                       _Float16* __restrict__ yg16)
{
  const int g = blockIdx.x * 256 + threadIdx.x;   // < 32768 = 256 tok * 128 ch
  const int d = g & 127, t = g >> 7;
  float Ad[DSTATE], h[DSTATE];
#pragma unroll
  for (int s = 0; s < DSTATE; ++s) { Ad[s] = -__expf(A_log[d * 8 + s]); h[s] = 0.0f; }
  const float Dd = Dp[d];
  for (int l = 0; l < 32; ++l) {
    const int row = t * 32 + l;
    const float dtv = dt[(size_t)row * 128 + d];
    const float xv  = xs[(size_t)row * 128 + d];
    const float dtx = dtv * xv;
    const float* bc = xdbl + (size_t)row * 32;
    float y = 0.0f;
#pragma unroll
    for (int s = 0; s < DSTATE; ++s) {
      h[s] = h[s] * __expf(dtv * Ad[s]) + dtx * bc[8 + s];   // Bp at cols 8..15
      y += h[s] * bc[16 + s];                                // Cp at cols 16..23
    }
    y += Dd * xv;
    const float z = xz[(size_t)row * 256 + 128 + d];
    y *= z / (1.0f + __expf(-z));                            // * silu(z)
    yg16[(size_t)row * 128 + d] = (_Float16)y;
  }
}

// ---------------- K7: feats = yg @ out_proj^T (8192x128, K=128) ------------
__global__ void k_gemm_outproj(const _Float16* __restrict__ A,
                               const _Float16* __restrict__ W,
                               _Float16* __restrict__ feats16)
{
  const int tile = blockIdx.x * 8 + (threadIdx.x >> 5);
  const int m0 = (tile >> 3) << 4, n0 = (tile & 7) << 4;
  float8 acc = wmma_tile_f16(A, W, m0, n0, 128, 128, 128);
  const int lane = threadIdx.x & 31;
  const int col = n0 + (lane & 15), rbase = m0 + (lane >> 4) * 8;
#pragma unroll
  for (int r = 0; r < 8; ++r)
    feats16[(size_t)(rbase + r) * 128 + col] = (_Float16)acc[r];
}

// ---------------- K8: heads GEMM (N=80 padded) + bias + transpose scatter --
__global__ void k_gemm_heads(const _Float16* __restrict__ A,
                             const _Float16* __restrict__ W,
                             const float* __restrict__ det_b,
                             const float* __restrict__ id_b,
                             float* __restrict__ out)
{
  const int tile = blockIdx.x * 8 + (threadIdx.x >> 5);
  const int m0 = (tile / 5) << 4, n0 = (tile % 5) << 4;
  float8 acc = wmma_tile_f16(A, W, m0, n0, 128, 128, 128);
  const int lane = threadIdx.x & 31;
  const int n = n0 + (lane & 15), rbase = m0 + (lane >> 4) * 8;
#pragma unroll
  for (int r = 0; r < 8; ++r) {
    const int row = rbase + r, t = row >> 5, b = row & 31;
    if (n < 5)
      out[(size_t)b * 1280 + n * 256 + t] = acc[r] + det_b[n];
    else if (n < 69)
      out[40960 + (size_t)b * 16384 + (n - 5) * 256 + t] = acc[r] + id_b[n - 5];
  }
}

// ---------------------------------------------------------------------------
static inline size_t align256(size_t x) { return (x + 255) & ~(size_t)255; }

extern "C" void kernel_launch(void* const* d_in, const int* in_sizes, int n_in,
                              void* d_out, int out_size, void* d_ws, size_t ws_size,
                              hipStream_t stream)
{
  const float* x         = (const float*)d_in[0];
  const float* conv1_w   = (const float*)d_in[1];
  const float* conv1_b   = (const float*)d_in[2];
  const float* in_proj_w = (const float*)d_in[3];
  const float* conv1d_w  = (const float*)d_in[4];
  const float* conv1d_b  = (const float*)d_in[5];
  const float* x_proj_w  = (const float*)d_in[6];
  const float* dt_proj_w = (const float*)d_in[7];
  const float* dt_proj_b = (const float*)d_in[8];
  const float* A_log     = (const float*)d_in[9];
  const float* D_param   = (const float*)d_in[10];
  const float* out_proj_w= (const float*)d_in[11];
  const float* det_w     = (const float*)d_in[12];
  const float* det_b     = (const float*)d_in[13];
  const float* id_w      = (const float*)d_in[14];
  const float* id_b      = (const float*)d_in[15];
  float* out = (float*)d_out;

  // ---- workspace carve-up (~27 MB) ----
  uintptr_t p = (uintptr_t)d_ws;
  auto takeH = [&](size_t n) { _Float16* q = (_Float16*)p; p = (uintptr_t)align256(p + n * 2); return q; };
  auto takeF = [&](size_t n) { float*    q = (float*)p;    p = (uintptr_t)align256(p + n * 4); return q; };
  _Float16* w_in   = takeH(256 * 128);
  _Float16* w_xp   = takeH(32 * 128);
  _Float16* w_dt   = takeH(128 * 32);
  _Float16* w_out  = takeH(128 * 128);
  _Float16* w_hd   = takeH(80 * 128);
  _Float16* u16    = takeH((size_t)ROWS * 128);
  float*    xz     = takeF((size_t)ROWS * 256);
  float*    xs     = takeF((size_t)ROWS * 128);
  _Float16* xs16   = takeH((size_t)ROWS * 128);
  float*    xdbl   = takeF((size_t)ROWS * 32);
  _Float16* xdbl16 = takeH((size_t)ROWS * 32);
  float*    dtbuf  = takeF((size_t)ROWS * 128);
  _Float16* yg16   = takeH((size_t)ROWS * 128);
  _Float16* f16buf = takeH((size_t)ROWS * 128);
  (void)ws_size; (void)in_sizes; (void)n_in; (void)out_size;

  // K0: weight pack (67584 elems / 256 = 264 blocks)
  k_pack_weights<<<264, 256, 0, stream>>>(in_proj_w, x_proj_w, dt_proj_w,
                                          out_proj_w, det_w, id_w,
                                          w_in, w_xp, w_dt, w_out, w_hd);
  // K1: fused stem+pool -> u16 (8192 blocks, one per (token,image))
  k_stem_pool<<<ROWS, 256, 0, stream>>>(x, conv1_w, conv1_b, u16);
  // K2: in_proj, 8192 tiles -> 1024 blocks
  k_gemm_inproj<<<1024, 256, 0, stream>>>(u16, w_in, xz);
  // K3: depthwise conv + silu, 1M elems -> 4096 blocks
  k_dwconv_silu<<<4096, 256, 0, stream>>>(xz, conv1d_w, conv1d_b, xs, xs16);
  // K4: x_proj, 1024 tiles -> 128 blocks
  k_gemm_xproj<<<128, 256, 0, stream>>>(xs16, w_xp, xdbl, xdbl16);
  // K5: dt_proj + softplus, 4096 tiles -> 512 blocks
  k_gemm_dtproj<<<512, 256, 0, stream>>>(xdbl16, w_dt, dt_proj_b, dtbuf);
  // K6: selective scan + gate, 32768 threads -> 128 blocks
  k_scan<<<128, 256, 0, stream>>>(dtbuf, xs, xdbl, xz, A_log, D_param, yg16);
  // K7: out_proj, 4096 tiles -> 512 blocks
  k_gemm_outproj<<<512, 256, 0, stream>>>(yg16, w_out, f16buf);
  // K8: heads + bias + scatter, 2560 tiles -> 320 blocks
  k_gemm_heads<<<320, 256, 0, stream>>>(f16buf, w_hd, det_b, id_b, out);
}
